// SAGEModel_47553877901463
// MI455X (gfx1250) — matrix operand order repaired and checked by reading, not compiled, not run on hardware
//
#include <hip/hip_runtime.h>
#include <hip/hip_bf16.h>

#define D_DIM 128
#define TILE_M 16
#define F_STRIDE 132   // f32 LDS row stride (pad kills bank conflicts)
#define B_STRIDE 136   // bf16 LDS row stride: lane m -> bank (4m + k/2)%64, conflict-free

typedef __attribute__((ext_vector_type(16))) __bf16 v16bf;
typedef __attribute__((ext_vector_type(8)))  __bf16 v8bf;
typedef __attribute__((ext_vector_type(4)))  __bf16 v4bf;
typedef __attribute__((ext_vector_type(8)))  float  v8f;

// ---------------------------------------------------------------- utilities

__global__ __launch_bounds__(256)
void deg_kernel(const int* __restrict__ erow, float* __restrict__ deg, int E) {
    int i = blockIdx.x * 256 + threadIdx.x;
    if (i < E) atomicAdd(&deg[erow[i]], 1.0f);
}

__global__ __launch_bounds__(256)
void invdeg_kernel(float* __restrict__ deg, int n) {
    int i = blockIdx.x * 256 + threadIdx.x;
    if (i < n) deg[i] = 1.0f / fmaxf(deg[i], 1.0f);
}

// Split fp32 weight into bf16 hi + bf16 lo (x ~= hi + lo), layout preserved.
__global__ __launch_bounds__(256)
void split_bf16_kernel(const float* __restrict__ w,
                       __bf16* __restrict__ hi, __bf16* __restrict__ lo, int n) {
    int i = blockIdx.x * 256 + threadIdx.x;
    if (i < n) {
        float f  = w[i];
        __bf16 h = (__bf16)f;
        hi[i] = h;
        lo[i] = (__bf16)(f - (float)h);
    }
}

// neigh[row] += h[col]  (UNSCALED; 1/deg applied later per destination row).
// One wave (32 lanes) per edge, float4 per lane.
__global__ __launch_bounds__(256)
void spmm_kernel(const float* __restrict__ h,
                 const int* __restrict__ erow, const int* __restrict__ ecol,
                 float* __restrict__ neigh, int E) {
    int e = blockIdx.x * 8 + (threadIdx.x >> 5);
    if (e >= E) return;
    int lane = threadIdx.x & 31;
    int r = erow[e], c = ecol[e];
    float4 hv = ((const float4*)(h + (size_t)c * D_DIM))[lane];
    float* dst = neigh + (size_t)r * D_DIM + lane * 4;
    atomicAdd(dst + 0, hv.x);
    atomicAdd(dst + 1, hv.y);
    atomicAdd(dst + 2, hv.z);
    atomicAdd(dst + 3, hv.w);
}

// ---------------------------------------------------------------- WMMA GEMM

__device__ __forceinline__ v8f wmma_bf16(v16bf a, v16bf b, v8f c) {
    return __builtin_amdgcn_wmma_f32_16x16x32_bf16(
        /*neg_a=*/false, a, /*neg_b=*/false, b,
        /*c_mod=*/(short)0, c, /*reuse_a=*/false, /*reuse_b=*/false);
}

// A fragment (16x32 bf16 tile, lane m = row, kbase = ks + 16*(lane>>4)):
// elements 0..7 = K[kbase .. kbase+7], elements 8..15 = K[kbase+8 .. kbase+15]
// -> two contiguous ds_load_b128 from a bf16 row in LDS.
__device__ __forceinline__ v16bf load_frag_lds(const __bf16* base, int m, int kbase) {
    union { v16bf v; v8bf h[2]; } f;
    const __bf16* p = base + m * B_STRIDE + kbase;
    f.h[0] = *(const v8bf*)(p);
    f.h[1] = *(const v8bf*)(p + 8);
    return f.v;
}

__device__ __forceinline__ void split4(float4 v, v4bf* hi, v4bf* lo) {
    v4bf h, l;
    h[0] = (__bf16)v.x; l[0] = (__bf16)(v.x - (float)h[0]);
    h[1] = (__bf16)v.y; l[1] = (__bf16)(v.y - (float)h[1]);
    h[2] = (__bf16)v.z; l[2] = (__bf16)(v.z - (float)h[2]);
    h[3] = (__bf16)v.w; l[3] = (__bf16)(v.w - (float)h[3]);
    *hi = h; *lo = l;
}

// out = LN(A1@W1^T + [scale*A2raw@W2^T] + bias) -> PReLU -> [+A1 residual]
// Block: 16 rows x 128 cols; 8 waves, each owning one 16x16 tile.
// A rows split into bf16 hi/lo ONCE at staging; 3 independent WMMA acc chains.
// DUAL / RESID are compile-time: straight-line, branch-free main loop.
template <bool DUAL, bool RESID>
__global__ __launch_bounds__(256)
void gemm_ln_prelu_kernel(const float* __restrict__ A1, const float* __restrict__ A2,
                          const float* __restrict__ inv_deg,   // row scale for A2
                          const __bf16* __restrict__ B1hi, const __bf16* __restrict__ B1lo,
                          const __bf16* __restrict__ B2hi, const __bf16* __restrict__ B2lo,
                          const float* __restrict__ bias,
                          const float* __restrict__ ln_g, const float* __restrict__ ln_b,
                          const float* __restrict__ alpha_p,
                          float* __restrict__ out) {
    __shared__ float  As1 [TILE_M * F_STRIDE];   // f32 copy (residual rows)
    __shared__ float  Zs  [TILE_M * F_STRIDE];
    __shared__ __bf16 A1hi[TILE_M * B_STRIDE];
    __shared__ __bf16 A1lo[TILE_M * B_STRIDE];
    __shared__ __bf16 A2hi[TILE_M * B_STRIDE];
    __shared__ __bf16 A2lo[TILE_M * B_STRIDE];

    const int    tid  = threadIdx.x;
    const size_t row0 = (size_t)blockIdx.x * TILE_M;

    // Stage + split: 16x128 floats, 2 float4 per thread, bf16 hi/lo computed once.
#pragma unroll
    for (int i = 0; i < 2; ++i) {
        int f4 = tid + i * 256;      // 512 float4 total
        int r  = f4 >> 5;            // 32 float4 per row
        int c4 = f4 & 31;
        float4 v = ((const float4*)(A1 + (row0 + r) * D_DIM))[c4];
        *(float4*)&As1[r * F_STRIDE + c4 * 4] = v;
        v4bf hi, lo;
        split4(v, &hi, &lo);
        *(v4bf*)&A1hi[r * B_STRIDE + c4 * 4] = hi;
        *(v4bf*)&A1lo[r * B_STRIDE + c4 * 4] = lo;
        if (DUAL) {
            float  s  = inv_deg[row0 + r];            // per-destination-row scale
            float4 v2 = ((const float4*)(A2 + (row0 + r) * D_DIM))[c4];
            v2.x *= s; v2.y *= s; v2.z *= s; v2.w *= s;
            v4bf hi2, lo2;
            split4(v2, &hi2, &lo2);
            *(v4bf*)&A2hi[r * B_STRIDE + c4 * 4] = hi2;
            *(v4bf*)&A2lo[r * B_STRIDE + c4 * 4] = lo2;
        }
    }

    const int wave  = tid >> 5;
    const int lane  = tid & 31;
    const int m     = lane & 15;
    const int khalf = lane >> 4;
    const int n     = wave * 16 + m;   // output column of this lane

    // Warm the weight rows this lane will stream (global_prefetch_b8 path).
    __builtin_prefetch(B1hi + (size_t)n * D_DIM, 0, 2);
    __builtin_prefetch(B1lo + (size_t)n * D_DIM, 0, 2);
    if (DUAL) {
        __builtin_prefetch(B2hi + (size_t)n * D_DIM, 0, 2);
        __builtin_prefetch(B2lo + (size_t)n * D_DIM, 0, 2);
    }
    __syncthreads();

    // Three independent accumulator chains to avoid WMMA->WMMA RAW stalls.
    v8f cHH = {0.f,0.f,0.f,0.f,0.f,0.f,0.f,0.f};
    v8f cHL = cHH, cLH = cHH;

#pragma unroll
    for (int ks = 0; ks < D_DIM; ks += 32) {
        const int ka = ks + khalf * 16;   // A fragment K base
        const int kb = ks + khalf * 16;   // B fragment K base (16 contiguous k)
        v16bf a1h = load_frag_lds(A1hi, m, ka);
        v16bf a1l = load_frag_lds(A1lo, m, ka);
        v16bf b1h = *(const v16bf*)(B1hi + (size_t)n * D_DIM + kb);
        v16bf b1l = *(const v16bf*)(B1lo + (size_t)n * D_DIM + kb);
        if (DUAL) {
            v16bf a2h = load_frag_lds(A2hi, m, ka);
            v16bf a2l = load_frag_lds(A2lo, m, ka);
            v16bf b2h = *(const v16bf*)(B2hi + (size_t)n * D_DIM + kb);
            v16bf b2l = *(const v16bf*)(B2lo + (size_t)n * D_DIM + kb);
            cHH = wmma_bf16(a1h, b1h, cHH);
            cHL = wmma_bf16(a1h, b1l, cHL);
            cLH = wmma_bf16(a1l, b1h, cLH);
            cHH = wmma_bf16(a2h, b2h, cHH);
            cHL = wmma_bf16(a2h, b2l, cHL);
            cLH = wmma_bf16(a2l, b2h, cLH);
        } else {
            cHH = wmma_bf16(a1h, b1h, cHH);
            cHL = wmma_bf16(a1h, b1l, cHL);
            cLH = wmma_bf16(a1l, b1h, cLH);
        }
    }

    // Combine chains + bias, scatter tile to LDS (VGPR r -> row r + 8*khalf, col n).
    const float bcol = bias[n];
#pragma unroll
    for (int r = 0; r < 8; ++r)
        Zs[(r + khalf * 8) * F_STRIDE + n] = cHH[r] + cHL[r] + cLH[r] + bcol;
    __syncthreads();

    // LayerNorm + PReLU + residual: 16 threads per row, 8 cols each.
    const int rr = tid >> 4;
    const int cg = (tid & 15) * 8;
    const float* zrow = Zs + rr * F_STRIDE;
    float zin[8], s = 0.f, ss = 0.f;
#pragma unroll
    for (int j = 0; j < 8; ++j) {
        float z = zrow[cg + j];
        zin[j]  = z;
        s  += z;
        ss += z * z;
    }
#pragma unroll
    for (int off = 1; off < 16; off <<= 1) {
        s  += __shfl_xor(s,  off, 16);
        ss += __shfl_xor(ss, off, 16);
    }
    const float mean  = s * (1.0f / 128.0f);
    const float var   = fmaxf(ss * (1.0f / 128.0f) - mean * mean, 0.0f);
    const float rstd  = rsqrtf(var + 1e-5f);
    const float alpha = alpha_p[0];
    const float* res  = As1 + rr * F_STRIDE;

    float o[8];
#pragma unroll
    for (int j = 0; j < 8; ++j) {
        float zn = (zin[j] - mean) * rstd * ln_g[cg + j] + ln_b[cg + j];
        zn = (zn > 0.f) ? zn : alpha * zn;
        if (RESID) zn += res[cg + j];
        o[j] = zn;
    }
    float* orow = out + (row0 + rr) * D_DIM + cg;
    *(float4*)(orow)     = make_float4(o[0], o[1], o[2], o[3]);
    *(float4*)(orow + 4) = make_float4(o[4], o[5], o[6], o[7]);
}

// logits[i] = dot(z[i], W2) + b2 ; one wave per row.
__global__ __launch_bounds__(256)
void head_out_kernel(const float* __restrict__ z, const float* __restrict__ W2,
                     const float* __restrict__ b2, float* __restrict__ out, int n) {
    int row = blockIdx.x * 8 + (threadIdx.x >> 5);
    if (row >= n) return;
    int lane = threadIdx.x & 31;
    float4 zv = ((const float4*)(z + (size_t)row * D_DIM))[lane];
    float4 wv = ((const float4*)W2)[lane];
    float s = zv.x * wv.x + zv.y * wv.y + zv.z * wv.z + zv.w * wv.w;
#pragma unroll
    for (int off = 16; off > 0; off >>= 1) s += __shfl_xor(s, off, 32);
    if (lane == 0) out[row] = s + b2[0];
}

// ---------------------------------------------------------------- launch

extern "C" void kernel_launch(void* const* d_in, const int* in_sizes, int n_in,
                              void* d_out, int out_size, void* d_ws, size_t ws_size,
                              hipStream_t stream) {
    // setup_inputs() insertion order, depth-first:
    // 0:x, per block [W_self, b_self, W_neigh, ln_g, ln_b, alpha] x2,
    // head [W1, b1, ln_g, ln_b, alpha, W2, b2], then edge_row, edge_col.
    const float* x         = (const float*)d_in[0];
    const float* Wself[2]  = {(const float*)d_in[1], (const float*)d_in[7]};
    const float* bself[2]  = {(const float*)d_in[2], (const float*)d_in[8]};
    const float* Wneigh[2] = {(const float*)d_in[3], (const float*)d_in[9]};
    const float* lng[2]    = {(const float*)d_in[4], (const float*)d_in[10]};
    const float* lnb[2]    = {(const float*)d_in[5], (const float*)d_in[11]};
    const float* alp[2]    = {(const float*)d_in[6], (const float*)d_in[12]};
    const float* W1     = (const float*)d_in[13];
    const float* b1     = (const float*)d_in[14];
    const float* hlng   = (const float*)d_in[15];
    const float* hlnb   = (const float*)d_in[16];
    const float* halpha = (const float*)d_in[17];
    const float* W2     = (const float*)d_in[18];
    const float* b2     = (const float*)d_in[19];
    const int*   erow   = (const int*)d_in[20];
    const int*   ecol   = (const int*)d_in[21];

    const int    N  = in_sizes[0] / D_DIM;   // 100000 (divisible by 16)
    const int    E  = in_sizes[20];          // 1600000
    const size_t ND = (size_t)N * D_DIM;

    // Workspace layout.
    float*  h1    = (float*)d_ws;
    float*  h2    = h1 + ND;
    float*  neigh = h2 + ND;
    float*  degv  = neigh + ND;
    __bf16* wb    = (__bf16*)(degv + N);
    const int WSZ = D_DIM * D_DIM;
    const float* Wsrc[5] = {Wself[0], Wneigh[0], Wself[1], Wneigh[1], W1};
    __bf16* Bhi[5];
    __bf16* Blo[5];
    for (int i = 0; i < 5; ++i) {
        Bhi[i] = wb + (size_t)i * 2 * WSZ;
        Blo[i] = Bhi[i] + WSZ;
    }

    hipMemsetAsync(degv, 0, (size_t)N * sizeof(float), stream);
    hipMemsetAsync(neigh, 0, ND * sizeof(float), stream);
    for (int i = 0; i < 5; ++i)
        split_bf16_kernel<<<(WSZ + 255) / 256, 256, 0, stream>>>(Wsrc[i], Bhi[i], Blo[i], WSZ);
    deg_kernel<<<(E + 255) / 256, 256, 0, stream>>>(erow, degv, E);
    invdeg_kernel<<<(N + 255) / 256, 256, 0, stream>>>(degv, N);

    const int gemm_grid = N / TILE_M;
    const int spmm_grid = (E + 7) / 8;

    // Layer 0: x -> h1
    spmm_kernel<<<spmm_grid, 256, 0, stream>>>(x, erow, ecol, neigh, E);
    gemm_ln_prelu_kernel<true, true><<<gemm_grid, 256, 0, stream>>>(
        x, neigh, degv, Bhi[0], Blo[0], Bhi[1], Blo[1],
        bself[0], lng[0], lnb[0], alp[0], h1);

    // Layer 1: h1 -> h2
    hipMemsetAsync(neigh, 0, ND * sizeof(float), stream);
    spmm_kernel<<<spmm_grid, 256, 0, stream>>>(h1, erow, ecol, neigh, E);
    gemm_ln_prelu_kernel<true, true><<<gemm_grid, 256, 0, stream>>>(
        h1, neigh, degv, Bhi[2], Blo[2], Bhi[3], Blo[3],
        bself[1], lng[1], lnb[1], alp[1], h2);

    // Head MLP: h2 -> h1 (reuse), then logits
    gemm_ln_prelu_kernel<false, false><<<gemm_grid, 256, 0, stream>>>(
        h2, nullptr, nullptr, Bhi[4], Blo[4], nullptr, nullptr,
        b1, hlng, hlnb, halpha, h1);
    head_out_kernel<<<(N + 7) / 8, 256, 0, stream>>>(h1, W2, b2, (float*)d_out, N);
}
